// PAM_Module_86260123174056
// MI455X (gfx1250) — matrix operand order repaired and checked by reading, not compiled, not run on hardware
//
#include <hip/hip_runtime.h>
#include <hip/hip_bf16.h>

typedef __bf16 bfx;
typedef __attribute__((ext_vector_type(16))) __bf16 v16bf;
typedef __attribute__((ext_vector_type(8)))  __bf16 v8bf;
typedef __attribute__((ext_vector_type(8)))  float  v8f;

#define B_  8
#define C_  512
#define N_  4096
#define CQ_ 64

__device__ __forceinline__ v16bf join8(v8bf lo, v8bf hi) {
  v16bf r;
#pragma unroll
  for (int i = 0; i < 8; ++i) { r[i] = lo[i]; r[i + 8] = hi[i]; }
  return r;
}

// x [B,C,N] fp32 -> xT [B,N,C] bf16 (A-operand friendly layout)
__global__ void xpose_bf16(const float* __restrict__ x, bfx* __restrict__ xT) {
  size_t idx = (size_t)blockIdx.x * blockDim.x + threadIdx.x; // c fastest
  int c = (int)(idx % C_);
  size_t t = idx / C_;
  int n = (int)(t % N_);
  int b = (int)(t / N_);
  xT[idx] = (bfx)x[((size_t)b * C_ + c) * N_ + n];
}

__global__ void cvt_bf16_k(const float* __restrict__ s, bfx* __restrict__ d, int n) {
  int i = blockIdx.x * blockDim.x + threadIdx.x;
  if (i < n) d[i] = (bfx)s[i];
}

// out[b, m, o] = sum_c xT[b, m, c] * W[o, c] + bias[o]
// Wave computes a 32x64 tile: 2 A frags x 4 B frags (B reused twice, all B
// loads hoisted ahead of the WMMA group for load/math overlap).
// transposeOut == 0: out is [B, M, Od] bf16 ; == 1: out is [B, Od, M] bf16.
// grid: (M/32, Od/64, B), block: 32
__global__ void proj_wmma(const bfx* __restrict__ xT, const bfx* __restrict__ W,
                          const float* __restrict__ bias, bfx* __restrict__ out,
                          int Od, int transposeOut) {
  const int lane = threadIdx.x;
  const int col16 = lane & 15, half = lane >> 4;
  const int m0 = blockIdx.x * 32;
  const int o0 = blockIdx.y * 64;
  const int b  = blockIdx.z;

  const bfx* arow0 = xT + ((size_t)b * N_ + m0 + col16) * C_;
  const bfx* arow1 = arow0 + (size_t)16 * C_;
  const bfx* brow0 = W + (size_t)(o0 + col16) * C_;

  v8f acc[2][4];
  v8f zero = {};
#pragma unroll
  for (int g = 0; g < 2; ++g)
#pragma unroll
    for (int ct = 0; ct < 4; ++ct) acc[g][ct] = zero;

#pragma unroll 2
  for (int k = 0; k < C_; k += 32) {
    int ka = k + half * 16;
    v16bf a0 = *(const v16bf*)(arow0 + ka);
    v16bf a1 = *(const v16bf*)(arow1 + ka);
    v16bf bm[4];
#pragma unroll
    for (int ct = 0; ct < 4; ++ct)
      bm[ct] = *(const v16bf*)(brow0 + (size_t)ct * 16 * C_ + k + half * 16);
#pragma unroll
    for (int ct = 0; ct < 4; ++ct) {
      acc[0][ct] = __builtin_amdgcn_wmma_f32_16x16x32_bf16(false, a0, false, bm[ct],
                                                           (short)0, acc[0][ct], false, false);
      acc[1][ct] = __builtin_amdgcn_wmma_f32_16x16x32_bf16(false, a1, false, bm[ct],
                                                           (short)0, acc[1][ct], false, false);
    }
  }

  if (transposeOut) {
#pragma unroll
    for (int g = 0; g < 2; ++g)
#pragma unroll
      for (int ct = 0; ct < 4; ++ct) {
        int col = o0 + ct * 16 + col16;
        float bs = bias[col];
        v8bf pack;
#pragma unroll
        for (int r = 0; r < 8; ++r) pack[r] = (bfx)(acc[g][ct][r] + bs);
        *(v8bf*)(out + ((size_t)b * Od + col) * N_ + m0 + g * 16 + half * 8) = pack;
      }
  } else {
    bfx* obase = out + (size_t)b * N_ * Od;
#pragma unroll
    for (int g = 0; g < 2; ++g)
#pragma unroll
      for (int ct = 0; ct < 4; ++ct) {
        int col = o0 + ct * 16 + col16;
        float bs = bias[col];
#pragma unroll
        for (int r = 0; r < 8; ++r) {
          int row = m0 + g * 16 + r + half * 8;
          obase[(size_t)row * Od + col] = (bfx)(acc[g][ct][r] + bs);
        }
      }
  }
}

// Cooperative fused flash attention.
// Block = 128 threads (4 waves) covering 16 query rows x all 512 channels.
// Wave w computes S-subtile t=w (no S redundancy), softmax stats exchanged
// via LDS; P built once in LDS and consumed by all 4 waves; each wave owns
// 128 output channels with software-pipelined contiguous V loads.
// grid: (N/16, B), block: 128
__global__ void attn_wmma(const bfx* __restrict__ q, const bfx* __restrict__ k,
                          const bfx* __restrict__ vC, const float* __restrict__ x,
                          const float* __restrict__ gamma, float* __restrict__ out) {
  __shared__ bfx   P[16 * 64];
  __shared__ float Mbuf[4][16];
  __shared__ float Lbuf[4][16];

  const int tid  = threadIdx.x;
  const int wid  = tid >> 5;
  const int lane = tid & 31;
  const int col16 = lane & 15, half = lane >> 4;
  const int i0 = blockIdx.x * 16;
  const int c0 = wid * 128;
  const int b  = blockIdx.y;

  // Q tile (16 x 64) as two K=32 A-fragments, resident for whole loop
  v16bf aq[2];
  {
    const bfx* qrow = q + ((size_t)b * N_ + i0 + col16) * CQ_;
#pragma unroll
    for (int kk = 0; kk < 2; ++kk) {
      int ka = kk * 32 + half * 16;
      aq[kk] = *(const v16bf*)(qrow + ka);
    }
  }

  float m[8], l[8];
  v8f o[8];
  v8f zero = {};
#pragma unroll
  for (int r = 0; r < 8; ++r) { m[r] = -1e30f; l[r] = 0.f; }
#pragma unroll
  for (int ct = 0; ct < 8; ++ct) o[ct] = zero;

  for (int jc = 0; jc < N_; jc += 64) {
    // ---- S subtile (16 x 16 at columns jc + wid*16), 2 WMMAs per wave ----
    v8f s = zero;
    {
      const bfx* krow = k + ((size_t)b * N_ + jc + wid * 16 + col16) * CQ_;
      v16bf bm0 = *(const v16bf*)(krow + half * 16);
      v16bf bm1 = *(const v16bf*)(krow + 32 + half * 16);
      s = __builtin_amdgcn_wmma_f32_16x16x32_bf16(false, aq[0], false, bm0,
                                                  (short)0, s, false, false);
      s = __builtin_amdgcn_wmma_f32_16x16x32_bf16(false, aq[1], false, bm1,
                                                  (short)0, s, false, false);
    }

    // ---- partial row max of this subtile -> LDS ----
    float pmx[8];
#pragma unroll
    for (int r = 0; r < 8; ++r) {
      float mx = s[r];
      mx = fmaxf(mx, __shfl_xor(mx, 1));
      mx = fmaxf(mx, __shfl_xor(mx, 2));
      mx = fmaxf(mx, __shfl_xor(mx, 4));
      mx = fmaxf(mx, __shfl_xor(mx, 8));
      pmx[r] = mx;
    }
    if (col16 == 0) {
#pragma unroll
      for (int r = 0; r < 8; ++r) Mbuf[wid][half * 8 + r] = pmx[r];
    }
    __syncthreads();

    // ---- global row max, exp, P write, partial row sums -> LDS ----
    float sc_[8];
#pragma unroll
    for (int r = 0; r < 8; ++r) {
      int row = half * 8 + r;
      float mx = fmaxf(fmaxf(Mbuf[0][row], Mbuf[1][row]),
                       fmaxf(Mbuf[2][row], Mbuf[3][row]));
      float mn = fmaxf(m[r], mx);
      sc_[r] = __expf(m[r] - mn);
      m[r] = mn;
      float p = __expf(s[r] - mn);
      P[row * 64 + wid * 16 + col16] = (bfx)p;
      float rs = p;
      rs += __shfl_xor(rs, 1);
      rs += __shfl_xor(rs, 2);
      rs += __shfl_xor(rs, 4);
      rs += __shfl_xor(rs, 8);
      if (col16 == 0) Lbuf[wid][row] = rs;
    }
    __syncthreads();

    // ---- fold in row sums, rescale O ----
#pragma unroll
    for (int r = 0; r < 8; ++r) {
      int row = half * 8 + r;
      float rs = (Lbuf[0][row] + Lbuf[1][row]) + (Lbuf[2][row] + Lbuf[3][row]);
      l[r] = l[r] * sc_[r] + rs;
#pragma unroll
      for (int ct = 0; ct < 8; ++ct) o[ct][r] *= sc_[r];
    }

    // ---- full P (16 x 64) as A fragments ----
    v16bf ap0, ap1;
    {
      const bfx* prow = P + col16 * 64;
      ap0 = *(const v16bf*)(prow + half * 16);
      ap1 = *(const v16bf*)(prow + 32 + half * 16);
    }

    // ---- O += P @ V^T over this wave's 128 channels, pipelined V loads ----
    const bfx* vbase = vC + ((size_t)b * C_ + c0 + col16) * N_ + jc;
    v16bf bv0 = *(const v16bf*)(vbase + half * 16);
    v16bf bv1 = *(const v16bf*)(vbase + 32 + half * 16);
#pragma unroll
    for (int ct = 0; ct < 8; ++ct) {
      if (ct < 7) {
        const bfx* vn = vbase + (size_t)(ct + 1) * 16 * N_;
        v16bf nv0 = *(const v16bf*)(vn + half * 16);
        v16bf nv1 = *(const v16bf*)(vn + 32 + half * 16);
        o[ct] = __builtin_amdgcn_wmma_f32_16x16x32_bf16(false, ap0, false, bv0,
                                                        (short)0, o[ct], false, false);
        o[ct] = __builtin_amdgcn_wmma_f32_16x16x32_bf16(false, ap1, false, bv1,
                                                        (short)0, o[ct], false, false);
        bv0 = nv0; bv1 = nv1;
      } else {
        o[ct] = __builtin_amdgcn_wmma_f32_16x16x32_bf16(false, ap0, false, bv0,
                                                        (short)0, o[ct], false, false);
        o[ct] = __builtin_amdgcn_wmma_f32_16x16x32_bf16(false, ap1, false, bv1,
                                                        (short)0, o[ct], false, false);
      }
    }
    __syncthreads();
  }

  // ---- epilogue: out = gamma * O / l + x  (contiguous v8f along N) ----
  float g = gamma[0];
  float il[8];
#pragma unroll
  for (int r = 0; r < 8; ++r) il[r] = 1.0f / l[r];
#pragma unroll
  for (int ct = 0; ct < 8; ++ct) {
    int cc = c0 + ct * 16 + col16;
    size_t base = ((size_t)b * C_ + cc) * N_ + i0 + half * 8;
    v8f xv = *(const v8f*)(x + base);
    v8f res;
#pragma unroll
    for (int r = 0; r < 8; ++r) res[r] = g * (o[ct][r] * il[r]) + xv[r];
    *(v8f*)(out + base) = res;
  }
}

extern "C" void kernel_launch(void* const* d_in, const int* in_sizes, int n_in,
                              void* d_out, int out_size, void* d_ws, size_t ws_size,
                              hipStream_t stream) {
  const float* x     = (const float*)d_in[0];
  const float* Wq    = (const float*)d_in[1];
  const float* bq    = (const float*)d_in[2];
  const float* Wk    = (const float*)d_in[3];
  const float* bk    = (const float*)d_in[4];
  const float* Wv    = (const float*)d_in[5];
  const float* bv    = (const float*)d_in[6];
  const float* gamma = (const float*)d_in[7];
  float* out = (float*)d_out;

  char* ws = (char*)d_ws;
  bfx* xT  = (bfx*)(ws);                          // B*N*C  bf16 = 32 MB  [B,N,C]
  bfx* vCb = (bfx*)(ws + (size_t)33554432);       // B*C*N  bf16 = 32 MB  [B,C,N]
  bfx* qb  = (bfx*)(ws + (size_t)67108864);       // B*N*Cq bf16 =  4 MB  [B,N,Cq]
  bfx* kb  = (bfx*)(ws + (size_t)71303168);       // B*N*Cq bf16 =  4 MB  [B,N,Cq]
  bfx* Wqb = (bfx*)(ws + (size_t)75497472);       // 64 KB
  bfx* Wkb = (bfx*)(ws + (size_t)75563008);       // 64 KB
  bfx* Wvb = (bfx*)(ws + (size_t)75628544);       // 512 KB

  int totalX = B_ * N_ * C_;
  xpose_bf16<<<totalX / 256, 256, 0, stream>>>(x, xT);
  cvt_bf16_k<<<(CQ_ * C_ + 255) / 256, 256, 0, stream>>>(Wq, Wqb, CQ_ * C_);
  cvt_bf16_k<<<(CQ_ * C_ + 255) / 256, 256, 0, stream>>>(Wk, Wkb, CQ_ * C_);
  cvt_bf16_k<<<(C_ * C_ + 255) / 256, 256, 0, stream>>>(Wv, Wvb, C_ * C_);

  proj_wmma<<<dim3(N_ / 32, CQ_ / 64, B_), 32, 0, stream>>>(xT, Wqb, bq, qb, CQ_, 0);
  proj_wmma<<<dim3(N_ / 32, CQ_ / 64, B_), 32, 0, stream>>>(xT, Wkb, bk, kb, CQ_, 0);
  proj_wmma<<<dim3(N_ / 32, C_  / 64, B_), 32, 0, stream>>>(xT, Wvb, bv, vCb, C_, 1);

  attn_wmma<<<dim3(N_ / 16, B_), 128, 0, stream>>>(qb, kb, vCb, x, gamma, out);
}
